// OCDIB_25434796327373
// MI455X (gfx1250) — compile-verified, matrix operand order checked by reading, not run on hardware
//
#include <hip/hip_runtime.h>
#include <hip/hip_bf16.h>

#define NN 50000
#define EE 800000
#define FH 128
#define KD 256           // K*D
#define CO2 512          // 2*K*D
#define MTILES (NN / 16) // 3125

typedef __attribute__((ext_vector_type(16))) __bf16 v16bf;
typedef __attribute__((ext_vector_type(16))) unsigned short v16us;
typedef __attribute__((ext_vector_type(8))) float v8f;
typedef __attribute__((ext_vector_type(4))) unsigned int u32x4;
typedef __attribute__((ext_vector_type(4))) int i32x4;
typedef __attribute__((ext_vector_type(8))) int i32x8;

#if __has_builtin(__builtin_amdgcn_tensor_load_to_lds)
#define VGAE_HAS_TDM 1
#if defined(__clang_major__) && __clang_major__ >= 23
#define VGAE_TDM_LOAD(g0, g1, g2, g3) \
  __builtin_amdgcn_tensor_load_to_lds((g0), (g1), (g2), (g3), (i32x8){0,0,0,0,0,0,0,0}, 0)
#else
#define VGAE_TDM_LOAD(g0, g1, g2, g3) \
  __builtin_amdgcn_tensor_load_to_lds((g0), (g1), (g2), (g3), 0)
#endif
#else
#define VGAE_HAS_TDM 0
#endif

__device__ __forceinline__ void vgae_wait_tensorcnt0() {
#if __has_builtin(__builtin_amdgcn_s_wait_tensorcnt)
  __builtin_amdgcn_s_wait_tensorcnt(0);
#else
  asm volatile("s_wait_tensorcnt 0x0" ::: "memory");
#endif
}

__device__ __forceinline__ unsigned short vgae_f2bf(float f) {
  unsigned int u = __float_as_uint(f);
  u += 0x7fffu + ((u >> 16) & 1u);        // round-to-nearest-even
  return (unsigned short)(u >> 16);
}

#if VGAE_HAS_TDM
// TDM descriptor group 0: count=1, lds byte address, 57-bit global addr, type=2.
__device__ __forceinline__ u32x4 vgae_tdm_g0(unsigned lds_addr, unsigned long long ga) {
  u32x4 g;
  g[0] = 1u;                                                   // count=1, user mode
  g[1] = lds_addr;                                             // lds_addr [63:32]
  g[2] = (unsigned)(ga & 0xffffffffull);                       // global_addr[31:0]
  g[3] = (unsigned)((ga >> 32) & 0x01ffffffull) | (2u << 30);  // addr[56:32] | type=2
  return g;
}
// Group 1 (2-D tile): data_size log2, tensor dims, tile dims, dim0 stride (elements).
__device__ __forceinline__ i32x8 vgae_tdm_g1(unsigned dsz, unsigned td0, unsigned td1,
                                             unsigned tile0, unsigned tile1,
                                             unsigned long long str0) {
  i32x8 g;
  g[0] = (int)(dsz << 16);                                   // wg_mask=0, flags=0
  g[1] = (int)((td0 & 0xffffu) << 16);                       // bar_addr=0 | td0[15:0]
  g[2] = (int)((td0 >> 16) | ((td1 & 0xffffu) << 16));       // td0[31:16] | td1[15:0]
  g[3] = (int)((td1 >> 16) | (tile0 << 16));                 // td1[31:16] | tile_dim0
  g[4] = (int)(tile1 & 0xffffu);                             // tile_dim1 | tile_dim2=0
  g[5] = (int)(str0 & 0xffffffffull);                        // dim0_stride[31:0]
  g[6] = (int)((str0 >> 32) & 0xffffull);                    // dim0_stride[47:32]
  g[7] = 0;
  return g;
}
#endif

// ---------------- small utility kernels ----------------

__global__ void vgae_fill(float* __restrict__ p, float v, int n) {
  int i = blockIdx.x * blockDim.x + threadIdx.x;
  if (i < n) p[i] = v;
}

__global__ void vgae_deg_edges(const int* __restrict__ ei, float* __restrict__ deg) {
  int e = blockIdx.x * blockDim.x + threadIdx.x;
  if (e < EE) atomicAdd(&deg[ei[EE + e]], 1.0f);
}

__global__ void vgae_rsqrt_ip(float* __restrict__ d, int n) {
  int i = blockIdx.x * blockDim.x + threadIdx.x;
  if (i < n) d[i] = rsqrtf(d[i]);
}

// Transposed bf16 repack: o[c*128 + k] = bf16(W[k*128 + c])   (128x128)
__global__ void vgae_convw_t(const float* __restrict__ w, unsigned short* __restrict__ o) {
  int i = blockIdx.x * blockDim.x + threadIdx.x;
  if (i >= FH * FH) return;
  int c = i >> 7, k = i & 127;
  o[i] = vgae_f2bf(w[k * FH + c]);
}

// Transposed repack of [Wmu|Wlv]: o[c*128 + h], c<256 -> Wmu[k=c/64][h][d=c%64].
__global__ void vgae_convwcat_t(const float* __restrict__ Wmu, const float* __restrict__ Wlv,
                                unsigned short* __restrict__ o) {
  int i = blockIdx.x * blockDim.x + threadIdx.x;
  if (i >= CO2 * FH) return;
  int c = i >> 7, h = i & 127;
  float v;
  if (c < KD) {
    int k = c >> 6, d = c & 63;
    v = Wmu[(k * FH + h) * 64 + d];
  } else {
    int c2 = c - KD, k = c2 >> 6, d = c2 & 63;
    v = Wlv[(k * FH + h) * 64 + d];
  }
  o[i] = vgae_f2bf(v);
}

// ---------------- WMMA GEMM: C(N x Co) = A(N x 128) * W(128 x Co) ----------------
// Block = 256 threads (8 waves) handles a 128-row slab x all Co columns.
// TDM stages W^T (Co x 128 bf16) and the A slab (128 x 128 f32) into LDS;
// fragments are assembled from LDS with contiguous b128 reads per the CDNA5
// 16-bit A 16x32 / f32 C/D 16x16 register layouts (ISA 7.12.2).

__global__ void vgae_gemm_wmma(const float* __restrict__ A,
                               const unsigned short* __restrict__ WbT, // Co x 128 bf16
                               float* __restrict__ C, int Co) {
  extern __shared__ char smem[];
  unsigned short* Wlds = (unsigned short*)smem;                  // Co*128 bf16
  float* Alds = (float*)(smem + (size_t)Co * FH * 2);            // 128*128 f32

  const int rowBase = blockIdx.x * 128;
  const int rows = (NN - rowBase < 128) ? (NN - rowBase) : 128;

#if VGAE_HAS_TDM
  if (threadIdx.x < 32) {   // wave 0 issues both DMAs (EXEC ignored by TDM)
    u32x4 wg0 = vgae_tdm_g0((unsigned)(unsigned long long)Wlds,
                            (unsigned long long)WbT);
    i32x8 wg1 = vgae_tdm_g1(/*2B*/1, FH, (unsigned)Co, FH, (unsigned)Co, FH);
    i32x4 z = {0, 0, 0, 0};
    VGAE_TDM_LOAD(wg0, wg1, z, z);
    u32x4 ag0 = vgae_tdm_g0((unsigned)(unsigned long long)Alds,
                            (unsigned long long)(A + (size_t)rowBase * FH));
    i32x8 ag1 = vgae_tdm_g1(/*4B*/2, FH, (unsigned)rows, FH, (unsigned)rows, FH);
    VGAE_TDM_LOAD(ag0, ag1, z, z);
    vgae_wait_tensorcnt0();
  }
#else
  for (int i = threadIdx.x; i < Co * FH; i += blockDim.x) Wlds[i] = WbT[i];
  for (int i = threadIdx.x; i < rows * FH; i += blockDim.x) Alds[i] = A[(size_t)rowBase * FH + i];
#endif
  __syncthreads();

  const int wave = threadIdx.x >> 5, lane = threadIdx.x & 31;
  if (rowBase + wave * 16 >= NN) return;          // wave-uniform (WMMA needs full EXEC)
  const int half = lane >> 4, l15 = lane & 15;
  const int rowL = wave * 16 + l15;               // A row inside the slab

  // A fragments for the 4 K-chunks, built once and reused across all tn tiles.
  // element i of lane -> K = 16*(i/8) + 8*half + (i%8)  => two contiguous 8-runs.
  v16us afrag[4];
#pragma unroll
  for (int kc = 0; kc < 4; ++kc) {
#pragma unroll
    for (int g = 0; g < 2; ++g) {
      const float* ap = &Alds[rowL * FH + kc * 32 + g * 16 + half * 8];
#pragma unroll
      for (int j = 0; j < 8; ++j) afrag[kc][g * 8 + j] = vgae_f2bf(ap[j]);
    }
  }

  const int ntn = Co >> 4;
  for (int tn = 0; tn < ntn; ++tn) {
    const unsigned short* wrow = &Wlds[(tn * 16 + l15) * FH]; // B col = l15 of tile
    v8f acc = {};
#pragma unroll
    for (int kc = 0; kc < 4; ++kc) {
      v16us bu;
#pragma unroll
      for (int g = 0; g < 2; ++g) {
        const unsigned short* bp = &wrow[kc * 32 + g * 16 + half * 8]; // 16B contiguous
#pragma unroll
        for (int j = 0; j < 8; ++j) bu[g * 8 + j] = bp[j];
      }
      acc = __builtin_amdgcn_wmma_f32_16x16x32_bf16(
          false, __builtin_bit_cast(v16bf, afrag[kc]),
          false, __builtin_bit_cast(v16bf, bu),
          (short)0, acc, false, false);
    }
#pragma unroll
    for (int v = 0; v < 8; ++v)   // f32 C/D: VGPR v -> M = v + 8*half, N = l15
      C[(size_t)(rowBase + wave * 16 + v + half * 8) * Co + tn * 16 + l15] = acc[v];
  }
}

// ---------------- edge propagation: out[dst] += in[src] * dinv[src]*dinv[dst] ----------------

__global__ void vgae_prop_edges(const float* __restrict__ in, const int* __restrict__ ei,
                                const float* __restrict__ dinv, float* __restrict__ out) {
  int t = blockIdx.x * blockDim.x + threadIdx.x;
  int e = t >> 5, lane = t & 31;
  if (e >= EE) return;
  int s = ei[e], d = ei[EE + e];
  float nrm = dinv[s] * dinv[d];
  float4 v = ((const float4*)(in + (long)s * FH))[lane];
  float* o = out + (long)d * FH + lane * 4;
  atomicAdd(o + 0, v.x * nrm);
  atomicAdd(o + 1, v.y * nrm);
  atomicAdd(o + 2, v.z * nrm);
  atomicAdd(o + 3, v.w * nrm);
}

// self-loop + bias + PReLU fused:  out = prelu(t + sl*dinv^2 + b, a)
__global__ void vgae_combine(const float* __restrict__ t, const float* __restrict__ sl,
                             const float* __restrict__ dinv, const float* __restrict__ bias,
                             const float* __restrict__ alpha, float* __restrict__ out) {
  int i = blockIdx.x * blockDim.x + threadIdx.x;
  if (i >= NN * FH) return;
  int n = i >> 7, f = i & 127;
  float di = dinv[n];
  float v = t[i] + sl[i] * di * di + (bias ? bias[f] : 0.0f);
  float a = alpha ? alpha[0] : 1.0f;
  out[i] = v >= 0.0f ? v : a * v;
}

// ---------------- per-node head: mu / logvar / hk / r ----------------

__global__ void vgae_finalize(const float* __restrict__ raw, const float* __restrict__ bmu,
                              const float* __restrict__ amu, const float* __restrict__ blv,
                              const float* __restrict__ alv, const float* __restrict__ Wr,
                              const float* __restrict__ br, const float* __restrict__ eps,
                              float* __restrict__ out) {
  int tid = threadIdx.x;
  int wave = tid >> 5, lane = tid & 31;
  int n = blockIdx.x * (blockDim.x >> 5) + wave;
  if (n >= NN) return;
  const long base_lv = (long)NN * FH;
  const long base_mu = base_lv + (long)NN * KD;
  const long base_hk = base_mu + (long)NN * KD;
  const long base_r  = base_hk + (long)NN * KD;
  const float* r0 = raw + (long)n * CO2;
  float racc = 0.0f;
  int kk = lane >> 3;                       // k of this 8-lane group
  float am = amu[kk], al = alv[kk];
#pragma unroll
  for (int j = 0; j < 8; ++j) {
    int c = lane * 8 + j;                   // c/64 == kk for all j
    float vm = r0[c] + bmu[c];
    vm = vm >= 0.0f ? vm : am * vm;                                 // mu (PReLU)
    float vl = r0[KD + c] + blv[c];
    vl = vl >= 0.0f ? vl : al * vl;
    vl = 1.0f / (1.0f + __expf(-vl));                               // logvar (sigmoid)
    float hk = eps[(long)n * KD + c] * __expf(0.5f * vl) + vm;      // reparameterize
    out[base_mu + (long)n * KD + c] = vm;
    out[base_lv + (long)n * KD + c] = vl;
    out[base_hk + (long)n * KD + c] = hk;
    racc += hk * Wr[c];
  }
  racc += __shfl_xor(racc, 1, 32);
  racc += __shfl_xor(racc, 2, 32);
  racc += __shfl_xor(racc, 4, 32);
  if ((lane & 7) == 0)
    out[base_r + (long)n * 4 + kk] = 1.0f / (1.0f + __expf(-(racc + br[kk])));
}

// ---------------- host-side orchestration ----------------

extern "C" void kernel_launch(void* const* d_in, const int* in_sizes, int n_in,
                              void* d_out, int out_size, void* d_ws, size_t ws_size,
                              hipStream_t stream) {
  const float* x   = (const float*)d_in[0];
  const int*   ei  = (const int*)  d_in[1];
  const float* W1  = (const float*)d_in[2];
  const float* b1  = (const float*)d_in[3];
  const float* a1  = (const float*)d_in[4];
  const float* W2  = (const float*)d_in[5];
  const float* b2  = (const float*)d_in[6];
  const float* a2  = (const float*)d_in[7];
  const float* Wmu = (const float*)d_in[8];
  const float* bmu = (const float*)d_in[9];
  const float* amu = (const float*)d_in[10];
  const float* Wlv = (const float*)d_in[11];
  const float* blv = (const float*)d_in[12];
  const float* alv = (const float*)d_in[13];
  const float* Wr  = (const float*)d_in[14];
  const float* br  = (const float*)d_in[15];
  const float* eps = (const float*)d_in[16];
  float* out = (float*)d_out;

  // workspace carve-up
  char* p = (char*)d_ws;
  float* dinv = (float*)p;            p += ((size_t)NN * 4 + 255) & ~(size_t)255;
  unsigned short* w1b = (unsigned short*)p;  p += (size_t)FH * FH * 2;   // W1^T bf16
  unsigned short* w2b = (unsigned short*)p;  p += (size_t)FH * FH * 2;   // W2^T bf16
  unsigned short* wcb = (unsigned short*)p;  p += (size_t)CO2 * FH * 2;  // [Wmu|Wlv]^T bf16
  float* bufA = (float*)p;            p += (size_t)NN * FH * 4;   // xw / t2
  float* bufB = (float*)p;            p += (size_t)NN * FH * 4;   // scatter accumulators
  float* bufC = (float*)p;            p += (size_t)NN * FH * 4;   // h / p
  float* raw  = (float*)p;            /* NN*CO2 floats */
  float* bufD = raw;                  // p-accumulator; dead before raw is written

  const int T = 256;
  const int gNF   = (NN * FH + T - 1) / T;
  const int gN    = (NN + T - 1) / T;
  const int gE    = (EE + T - 1) / T;
  const int gProp = (EE * 32) / T;
  const int gGemm = (NN + 127) / 128;                       // 391 slabs
  const size_t smem128 = (size_t)FH * FH * 2 + (size_t)FH * FH * 4;   //  96 KB
  const size_t smem512 = (size_t)CO2 * FH * 2 + (size_t)FH * FH * 4;  // 192 KB

  // degrees -> dinv
  vgae_fill<<<gN, T, 0, stream>>>(dinv, 1.0f, NN);          // self loops
  vgae_deg_edges<<<gE, T, 0, stream>>>(ei, dinv);
  vgae_rsqrt_ip<<<gN, T, 0, stream>>>(dinv, NN);

  // weight conversion to transposed bf16
  vgae_convw_t<<<(FH * FH + T - 1) / T, T, 0, stream>>>(W1, w1b);
  vgae_convw_t<<<(FH * FH + T - 1) / T, T, 0, stream>>>(W2, w2b);
  vgae_convwcat_t<<<(CO2 * FH + T - 1) / T, T, 0, stream>>>(Wmu, Wlv, wcb);

  // layer 1: h = prelu(prop(x@W1) + b1)
  vgae_gemm_wmma<<<gGemm, T, smem128, stream>>>(x, w1b, bufA, FH);
  vgae_fill<<<gNF, T, 0, stream>>>(bufB, 0.0f, NN * FH);
  vgae_prop_edges<<<gProp, T, 0, stream>>>(bufA, ei, dinv, bufB);
  vgae_combine<<<gNF, T, 0, stream>>>(bufB, bufA, dinv, b1, a1, bufC);

  // layer 2: hgcn = prelu(prop(h@W2) + b2) -> d_out[0 : N*128]
  vgae_gemm_wmma<<<gGemm, T, smem128, stream>>>(bufC, w2b, bufA, FH);
  vgae_fill<<<gNF, T, 0, stream>>>(bufB, 0.0f, NN * FH);
  vgae_prop_edges<<<gProp, T, 0, stream>>>(bufA, ei, dinv, bufB);
  vgae_combine<<<gNF, T, 0, stream>>>(bufB, bufA, dinv, b2, a2, out);

  // p = prop(hgcn)  (identity combine: null bias/alpha)
  vgae_fill<<<gNF, T, 0, stream>>>(bufD, 0.0f, NN * FH);
  vgae_prop_edges<<<gProp, T, 0, stream>>>(out, ei, dinv, bufD);
  vgae_combine<<<gNF, T, 0, stream>>>(bufD, out, dinv, nullptr, nullptr, bufC);

  // raw = p @ [Wmu | Wlv]   (N x 512)
  vgae_gemm_wmma<<<gGemm, T, smem512, stream>>>(bufC, wcb, raw, CO2);

  // head: logvar / mu / hk / r
  vgae_finalize<<<NN / 8, T, 0, stream>>>(raw, bmu, amu, blv, alv, Wr, br, eps, out);
}
